// PointNetSetAbstraction_29394756174196
// MI455X (gfx1250) — matrix operand order repaired and checked by reading, not compiled, not run on hardware
//
#include <hip/hip_runtime.h>
#include <hip/hip_bf16.h>

// PointNet SetAbstraction for MI455X (gfx1250, wave32).
// bf16 activations + v_wmma_f32_16x16x32_bf16, BN stats fused into GEMM epilogue.

typedef __attribute__((ext_vector_type(16))) __bf16 v16bf;
typedef __attribute__((ext_vector_type(8)))  float  v8f;

#define NB    16
#define NPTS  4096
#define NM    1024
#define NK    32
#define NCOLS (NB * NM * NK)   /* 524288 columns, col = ((b*NM+m)*NK + k) */
#define R2    0.16f
#define EPSBN 1e-5f

// ---------------- workspace layout (bytes, 256-aligned) ----------------
#define OFF_FPS  ((size_t)0)            /* int  [16*1024]            65536  */
#define OFF_NN   ((size_t)65536)        /* int  [16*1024*32]       2097152  */
#define OFF_W0   ((size_t)2162688)      /* bf16 [64*32]               4096  */
#define OFF_W1   ((size_t)2166784)      /* bf16 [64*64]               8192  */
#define OFF_W2   ((size_t)2174976)      /* bf16 [128*64]             16384  */
#define OFF_ST   ((size_t)2191360)      /* f32  3 layers * 512       6144   */
#define OFF_X0   ((size_t)2197504)      /* bf16 [NCOLS*32]        33554432  */
#define OFF_X1   ((size_t)35751936)     /* bf16 [NCOLS*64]        67108864  */
#define OFF_X2   ((size_t)102860800)    /* bf16 [NCOLS*64]        67108864  */
#define OFF_Y2   ((size_t)169969664)    /* bf16 [NCOLS*128]      134217728  */
/* total ~304 MB */

// ---------------------------------------------------------------------------
__global__ void zero_f32_kernel(float* p, int n) {
    int i = blockIdx.x * blockDim.x + threadIdx.x;
    if (i < n) p[i] = 0.f;
}

// pad/convert weights: Wb[cout, cinP] bf16, zero-padded beyond cin
__global__ void convw_kernel(const float* __restrict__ W, __bf16* __restrict__ Wb,
                             int cout, int cin, int cinP) {
    int i = blockIdx.x * blockDim.x + threadIdx.x;
    if (i >= cout * cinP) return;
    int co = i / cinP, ci = i - co * cinP;
    Wb[i] = (__bf16)(ci < cin ? W[co * cin + ci] : 0.f);
}

// ---------------------------------------------------------------------------
// Farthest point sampling: one block (256 threads) per batch; distances live
// in registers (16 pts/thread), argmax via shfl_xor + tiny LDS tree.
__global__ __launch_bounds__(256) void fps_kernel(const float* __restrict__ xyz,
                                                  int* __restrict__ fps_idx) {
    int b = blockIdx.x, t = threadIdx.x;
    const float* X = xyz + (size_t)b * 3 * NPTS;
    const float* Y = X + NPTS;
    const float* Z = Y + NPTS;
    float px[16], py[16], pz[16], dist[16];
#pragma unroll
    for (int j = 0; j < 16; ++j) {
        int n = t + j * 256;
        px[j] = X[n]; py[j] = Y[n]; pz[j] = Z[n];
        dist[j] = 1e10f;
    }
    __shared__ float swv[8];
    __shared__ int   swi[8];
    __shared__ int   sfar;
    int far = 0;
    for (int it = 0; it < NM; ++it) {
        if (t == 0) fps_idx[b * NM + it] = far;
        float sx = X[far], sy = Y[far], sz = Z[far];
        float bestv = -1.f; int besti = 0x7fffffff;
#pragma unroll
        for (int j = 0; j < 16; ++j) {
            float dx = px[j] - sx, dy = py[j] - sy, dz = pz[j] - sz;
            float d = dx * dx + dy * dy + dz * dz;
            dist[j] = fminf(dist[j], d);
            int n = t + j * 256;
            if (dist[j] > bestv || (dist[j] == bestv && n < besti)) { bestv = dist[j]; besti = n; }
        }
#pragma unroll
        for (int mm = 16; mm >= 1; mm >>= 1) {
            float ov = __shfl_xor(bestv, mm, 32);
            int   oi = __shfl_xor(besti, mm, 32);
            if (ov > bestv || (ov == bestv && oi < besti)) { bestv = ov; besti = oi; }
        }
        if ((t & 31) == 0) { swv[t >> 5] = bestv; swi[t >> 5] = besti; }
        __syncthreads();
        if (t == 0) {
            float bv = swv[0]; int bi = swi[0];
            for (int w = 1; w < 8; ++w)
                if (swv[w] > bv || (swv[w] == bv && swi[w] < bi)) { bv = swv[w]; bi = swi[w]; }
            sfar = bi;
        }
        __syncthreads();
        far = sfar;
    }
}

// out_xyz[b,c,m] = xyz[b,c,fps_idx[b,m]]  (written straight into d_out)
__global__ void sample_gather_kernel(const float* __restrict__ xyz,
                                     const int* __restrict__ fps_idx,
                                     float* __restrict__ out_xyz) {
    int i = blockIdx.x * blockDim.x + threadIdx.x;     // 16*3*1024
    if (i >= NB * 3 * NM) return;
    int m = i & (NM - 1);
    int bc = i >> 10;                                   // b*3 + c
    int b = bc / 3;
    out_xyz[i] = xyz[(size_t)bc * NPTS + fps_idx[b * NM + m]];
}

// k lowest-index in-radius neighbors per query; pad with first neighbor.
__global__ void ball_query_kernel(const float* __restrict__ xyz,
                                  const float* __restrict__ sample,
                                  int* __restrict__ nn_idx) {
    int q = blockIdx.x * blockDim.x + threadIdx.x;     // 16*1024
    if (q >= NB * NM) return;
    int b = q >> 10, m = q & (NM - 1);
    const float* X = xyz + (size_t)b * 3 * NPTS;
    const float* Y = X + NPTS;
    const float* Z = Y + NPTS;
    float qx = sample[(size_t)(b * 3 + 0) * NM + m];
    float qy = sample[(size_t)(b * 3 + 1) * NM + m];
    float qz = sample[(size_t)(b * 3 + 2) * NM + m];
    int* out = nn_idx + (size_t)q * NK;
    int cnt = 0, first = 0;
    for (int n = 0; n < NPTS && cnt < NK; ++n) {
        float dx = X[n] - qx, dy = Y[n] - qy, dz = Z[n] - qz;
        float d = dx * dx + dy * dy + dz * dz;
        if (d <= R2) {
            if (cnt == 0) first = n;
            out[cnt++] = n;
        }
    }
    for (; cnt < NK; ++cnt) out[cnt] = first;
}

// Build X0[col, 32] bf16: ch 0..2 = grouped xyz - sample, 3..5 = features, rest 0.
__global__ void group_kernel(const float* __restrict__ xyz,
                             const float* __restrict__ feat,
                             const float* __restrict__ sample,
                             const int* __restrict__ nn_idx,
                             __bf16* __restrict__ X0) {
    int col = blockIdx.x * blockDim.x + threadIdx.x;   // NCOLS
    if (col >= NCOLS) return;
    int bm = col >> 5;                                  // b*NM + m
    int m = bm & (NM - 1), b = bm >> 10;
    int n = nn_idx[col];
    union { uint4 q; __bf16 e[8]; } u;
#pragma unroll
    for (int c = 0; c < 3; ++c) {
        float g = xyz[(size_t)(b * 3 + c) * NPTS + n] - sample[(size_t)(b * 3 + c) * NM + m];
        u.e[c] = (__bf16)g;
        u.e[3 + c] = (__bf16)feat[(size_t)(b * 3 + c) * NPTS + n];
    }
    u.e[6] = (__bf16)0.f; u.e[7] = (__bf16)0.f;
    uint4* o = (uint4*)(X0 + (size_t)col * 32);
    uint4 z = make_uint4(0u, 0u, 0u, 0u);
    o[0] = u.q; o[1] = z; o[2] = z; o[3] = z;
}

// ---------------------------------------------------------------------------
// GEMM: Y[col, cout] (bf16, pre-BN) = W[cout, CINP] * X[col, CINP], with fused
// per-channel sum/sumsq (BN statistics) via shfl reduce + float atomics.
// One wave per 16(cout)x16(col) tile, 8 column-tiles per wave.
template <int CINP, int NKC>
__global__ __launch_bounds__(256) void gemm_bn_stats_kernel(
    const __bf16* __restrict__ Wb, const __bf16* __restrict__ Xin,
    __bf16* __restrict__ Yout, float* __restrict__ bnsum, float* __restrict__ bnsq,
    int cout, int couttiles) {
    int wave = (int)((blockIdx.x * blockDim.x + threadIdx.x) >> 5);
    int lane = threadIdx.x & 31;
    int colgrp = wave / couttiles;
    int ct = wave - colgrp * couttiles;
    int n = lane & 15, h = lane >> 4;
    union BV { uint4 q[2]; v16bf v; };

    // A (weights), ISA 16-bit A layout: lane M = n; halves h pick K sub-ranges.
    v16bf a[NKC];
    {
        const __bf16* wrow = Wb + (size_t)(ct * 16 + n) * CINP;
#pragma unroll
        for (int kc = 0; kc < NKC; ++kc) {
            BV av;
            const __bf16* wp = wrow + kc * 32 + h * 8;
            av.q[0] = *(const uint4*)(wp);       // K = h*8   .. h*8+7
            av.q[1] = *(const uint4*)(wp + 16);  // K = h*8+16.. h*8+23
            a[kc] = av.v;
        }
    }

    float lsum[8], lsq[8];
#pragma unroll
    for (int r = 0; r < 8; ++r) { lsum[r] = 0.f; lsq[r] = 0.f; }

    for (int t = 0; t < 8; ++t) {
        int col = (colgrp * 8 + t) * 16 + n;
        v8f acc = {};
#pragma unroll
        for (int kc = 0; kc < NKC; ++kc) {
            BV bv;   // B layout: lane col N = n, K = h*8..h*8+7, h*8+16..h*8+23
            const uint4* xp = (const uint4*)(Xin + (size_t)col * CINP + kc * 32 + h * 8);
            bv.q[0] = xp[0];
            bv.q[1] = xp[2];
            acc = __builtin_amdgcn_wmma_f32_16x16x32_bf16(
                false, a[kc], false, bv.v, (short)0, acc, false, false);
        }
        if (t < 7)
            __builtin_prefetch(Xin + (size_t)(col + 16) * CINP + h * 16, 0, 1);
        union { uint4 q; __bf16 e[8]; } ov;
#pragma unroll
        for (int r = 0; r < 8; ++r) {           // D: lane element r -> M = h*8+r, N = n
            float v = acc[r];
            ov.e[r] = (__bf16)v;
            lsum[r] += v;
            lsq[r]  += v * v;
        }
        *(uint4*)(Yout + (size_t)col * cout + ct * 16 + h * 8) = ov.q;
    }
    // reduce over the 16 columns held by each half-wave (xor 8..1 stays in-half)
#pragma unroll
    for (int r = 0; r < 8; ++r) {
#pragma unroll
        for (int mm = 8; mm >= 1; mm >>= 1) {
            lsum[r] += __shfl_xor(lsum[r], mm, 32);
            lsq[r]  += __shfl_xor(lsq[r],  mm, 32);
        }
    }
    if (n == 0) {
        int ch = ct * 16 + h * 8;
#pragma unroll
        for (int r = 0; r < 8; ++r) {
            atomicAdd(&bnsum[ch + r], lsum[r]);
            atomicAdd(&bnsq[ch + r],  lsq[r]);
        }
    }
}

// scale = g * rsqrt(var+eps), shift = bt - mu*scale (conv bias cancels in BN)
__global__ void finalize_stats_kernel(const float* __restrict__ bnsum,
                                      const float* __restrict__ bnsq,
                                      const float* __restrict__ g,
                                      const float* __restrict__ bt,
                                      float* __restrict__ scale,
                                      float* __restrict__ shift,
                                      int cout, float inv_n) {
    int c = blockIdx.x * blockDim.x + threadIdx.x;
    if (c >= cout) return;
    float mu = bnsum[c] * inv_n;
    float var = bnsq[c] * inv_n - mu * mu;
    float sc = g[c] * rsqrtf(var + EPSBN);
    scale[c] = sc;
    shift[c] = bt[c] - mu * sc;
}

// in-place BN + ReLU on bf16 activations, 8 channels per thread
__global__ void bn_relu_kernel(__bf16* __restrict__ Y,
                               const float* __restrict__ scale,
                               const float* __restrict__ shift,
                               int cout, int total8) {
    int i = blockIdx.x * blockDim.x + threadIdx.x;
    if (i >= total8) return;
    int c0 = (int)(((size_t)i * 8) % (size_t)cout);
    union { uint4 q; __bf16 e[8]; } u;
    u.q = *((const uint4*)Y + (size_t)i);
#pragma unroll
    for (int r = 0; r < 8; ++r) {
        float f = (float)u.e[r] * scale[c0 + r] + shift[c0 + r];
        u.e[r] = (__bf16)fmaxf(f, 0.f);
    }
    *((uint4*)Y + (size_t)i) = u.q;
}

// final: out[b,o,m] = max_k relu(scale[o]*Y2[col,o]+shift[o]); fuses layer-2 BN.
__global__ void final_max_kernel(const __bf16* __restrict__ Y2,
                                 const float* __restrict__ scale,
                                 const float* __restrict__ shift,
                                 float* __restrict__ outx) {
    int i = blockIdx.x * blockDim.x + threadIdx.x;     // (b*NM+m)*128 + o
    if (i >= NB * NM * 128) return;
    int o = i & 127;
    int bm = i >> 7;
    int m = bm & (NM - 1), b = bm >> 10;
    const __bf16* p = Y2 + (size_t)bm * NK * 128 + o;
    float sc = scale[o], sh = shift[o];
    float mx = 0.f;                                    // relu output is >= 0
#pragma unroll 4
    for (int k = 0; k < NK; ++k) {
        float f = (float)p[(size_t)k * 128] * sc + sh;
        mx = fmaxf(mx, fmaxf(f, 0.f));
    }
    outx[(size_t)(b * 128 + o) * NM + m] = mx;
}

// ---------------------------------------------------------------------------
extern "C" void kernel_launch(void* const* d_in, const int* in_sizes, int n_in,
                              void* d_out, int out_size, void* d_ws, size_t ws_size,
                              hipStream_t stream) {
    const float* xyz  = (const float*)d_in[0];
    const float* feat = (const float*)d_in[1];
    const float* W[3]  = { (const float*)d_in[2], (const float*)d_in[6],  (const float*)d_in[10] };
    const float* Gm[3] = { (const float*)d_in[4], (const float*)d_in[8],  (const float*)d_in[12] };
    const float* Bt[3] = { (const float*)d_in[5], (const float*)d_in[9],  (const float*)d_in[13] };
    float* out = (float*)d_out;                         // [16*3*1024] ++ [16*128*1024]
    char* ws = (char*)d_ws;

    int*    fps   = (int*)(ws + OFF_FPS);
    int*    nn    = (int*)(ws + OFF_NN);
    __bf16* Wb[3] = { (__bf16*)(ws + OFF_W0), (__bf16*)(ws + OFF_W1), (__bf16*)(ws + OFF_W2) };
    float*  st    = (float*)(ws + OFF_ST);              // per layer: 512 floats
    __bf16* X0 = (__bf16*)(ws + OFF_X0);
    __bf16* X1 = (__bf16*)(ws + OFF_X1);
    __bf16* X2 = (__bf16*)(ws + OFF_X2);
    __bf16* Y2 = (__bf16*)(ws + OFF_Y2);

    const int cin[3]  = { 6, 64, 64 };
    const int cinP[3] = { 32, 64, 64 };
    const int cout[3] = { 64, 64, 128 };
    float* bnsum[3], *bnsq[3], *scale[3], *shift[3];
    for (int l = 0; l < 3; ++l) {
        bnsum[l] = st + l * 512;
        bnsq[l]  = st + l * 512 + 128;
        scale[l] = st + l * 512 + 256;
        shift[l] = st + l * 512 + 384;
    }

    // 0) zero BN accumulators
    zero_f32_kernel<<<6, 256, 0, stream>>>(st, 1536);
    // 1) weights -> padded bf16
    for (int l = 0; l < 3; ++l) {
        int ne = cout[l] * cinP[l];
        convw_kernel<<<(ne + 255) / 256, 256, 0, stream>>>(W[l], Wb[l], cout[l], cin[l], cinP[l]);
    }
    // 2) FPS + gather sampled xyz (straight into d_out)
    fps_kernel<<<NB, 256, 0, stream>>>(xyz, fps);
    sample_gather_kernel<<<(NB * 3 * NM + 255) / 256, 256, 0, stream>>>(xyz, fps, out);
    // 3) ball query + grouping
    ball_query_kernel<<<(NB * NM + 255) / 256, 256, 0, stream>>>(xyz, out, nn);
    group_kernel<<<NCOLS / 256, 256, 0, stream>>>(xyz, feat, out, nn, X0);

    const float inv_n = 1.f / (float)NCOLS;
    const int colgrps = NCOLS / (16 * 8);               // 4096 groups of 128 cols

    // ---- layer 0: 6(pad32) -> 64
    gemm_bn_stats_kernel<32, 1><<<colgrps * 4 / 8, 256, 0, stream>>>(Wb[0], X0, X1, bnsum[0], bnsq[0], 64, 4);
    finalize_stats_kernel<<<1, 128, 0, stream>>>(bnsum[0], bnsq[0], Gm[0], Bt[0], scale[0], shift[0], 64, inv_n);
    bn_relu_kernel<<<(NCOLS * 64 / 8) / 256, 256, 0, stream>>>(X1, scale[0], shift[0], 64, NCOLS * 64 / 8);

    // ---- layer 1: 64 -> 64
    gemm_bn_stats_kernel<64, 2><<<colgrps * 4 / 8, 256, 0, stream>>>(Wb[1], X1, X2, bnsum[1], bnsq[1], 64, 4);
    finalize_stats_kernel<<<1, 128, 0, stream>>>(bnsum[1], bnsq[1], Gm[1], Bt[1], scale[1], shift[1], 64, inv_n);
    bn_relu_kernel<<<(NCOLS * 64 / 8) / 256, 256, 0, stream>>>(X2, scale[1], shift[1], 64, NCOLS * 64 / 8);

    // ---- layer 2: 64 -> 128 (BN+ReLU fused into the max kernel)
    gemm_bn_stats_kernel<64, 2><<<colgrps * 8 / 8, 256, 0, stream>>>(Wb[2], X2, Y2, bnsum[2], bnsq[2], 128, 8);
    finalize_stats_kernel<<<1, 128, 0, stream>>>(bnsum[2], bnsq[2], Gm[2], Bt[2], scale[2], shift[2], 128, inv_n);
    final_max_kernel<<<(NB * NM * 128) / 256, 256, 0, stream>>>(Y2, scale[2], shift[2], out);
}
